// PointNetSetAbstraction_5609227289178
// MI455X (gfx1250) — compile-verified, matrix operand order checked by reading, not run on hardware
//
#include <hip/hip_runtime.h>
#include <hip/hip_bf16.h>

typedef _Float16 f16_t;
typedef __attribute__((ext_vector_type(16))) _Float16 v16h;
typedef __attribute__((ext_vector_type(8)))  _Float16 v8h;
typedef __attribute__((ext_vector_type(8)))  float    v8f;

#define BATCH   16
#define NPTS    4096
#define DFEAT   64
#define NPOINT  1024
#define NSAMPLE 32
#define RADIUS2 (0.2f * 0.2f)
#define BN_EPS  1e-5f
#define ROWS    (BATCH * NPOINT * NSAMPLE)   // 524288
#define KPAD    96                            // 67 -> 96 (3 k-tiles of 32)
#define MTILES  (ROWS / 16)                   // 32768
#define GEMM_BLOCKS (MTILES / 8)              // 4096 (8 waves/block)

// ---------------- workspace layout (bytes) ----------------
#define STATS_OFF 0u                      // 3 layers * 256 f32 (sum[Nout], sq[Nout])
#define PAR_OFF   4096u                   // 3 layers * 256 f32 (sA[128], sB[128])
#define FPS_OFF   8192u                   // 16*1024 int
#define IDX_OFF   (8192u + 65536u)        // 16*1024*32 int -> ends 2170880
#define X_OFF     2170880u                // ROWS*96 f16 = 100663296 B
#define Y_OFF     102834176u              // ROWS*128 f16 = 134217728 B

// ---------------------------------------------------------------------------
// zero the BN statistics accumulators
__global__ void init_stats_kernel(float* stats) {
    int i = threadIdx.x;
    for (; i < 3 * 256; i += 256) stats[i] = 0.0f;
}

// ---------------------------------------------------------------------------
// Farthest point sampling: one block per batch, xyz + dist in LDS.
__global__ void fps_kernel(const float* __restrict__ xyz,
                           int* __restrict__ fps_idx,
                           float* __restrict__ new_xyz /* d_out part 0 */) {
    const int b   = blockIdx.x;
    const int tid = threadIdx.x;   // 256 threads
    __shared__ float sx[NPTS * 3];
    __shared__ float sdist[NPTS];
    __shared__ float rmax[256];
    __shared__ int   rarg[256];

    const float* src = xyz + (size_t)b * NPTS * 3;
    for (int i = tid; i < NPTS * 3; i += 256) sx[i] = src[i];
    for (int i = tid; i < NPTS;     i += 256) sdist[i] = 1e10f;
    __syncthreads();

    int far = 0;
    for (int it = 0; it < NPOINT; ++it) {
        const float cx = sx[far * 3 + 0];
        const float cy = sx[far * 3 + 1];
        const float cz = sx[far * 3 + 2];
        if (tid == 0) {
            fps_idx[b * NPOINT + it] = far;
            float* o = new_xyz + ((size_t)b * NPOINT + it) * 3;
            o[0] = cx; o[1] = cy; o[2] = cz;
        }
        float bestd = -1.0f; int besti = 0;
        #pragma unroll 4
        for (int j = 0; j < NPTS / 256; ++j) {
            const int p = tid * (NPTS / 256) + j;   // contiguous chunk: index order preserved
            const float dx = sx[p * 3 + 0] - cx;
            const float dy = sx[p * 3 + 1] - cy;
            const float dz = sx[p * 3 + 2] - cz;
            const float d  = dx * dx + dy * dy + dz * dz;
            const float nd = fminf(sdist[p], d);
            sdist[p] = nd;
            if (nd > bestd) { bestd = nd; besti = p; }
        }
        rmax[tid] = bestd; rarg[tid] = besti;
        __syncthreads();
        for (int s = 128; s > 0; s >>= 1) {
            if (tid < s) {
                const float od = rmax[tid + s]; const int oi = rarg[tid + s];
                if (od > rmax[tid] || (od == rmax[tid] && oi < rarg[tid])) {
                    rmax[tid] = od; rarg[tid] = oi;
                }
            }
            __syncthreads();
        }
        far = rarg[0];
        __syncthreads();
    }
}

// ---------------------------------------------------------------------------
// Ball query: scan points in index order, keep first NSAMPLE inside radius.
__global__ void ball_kernel(const float* __restrict__ xyz,
                            const float* __restrict__ new_xyz,
                            int* __restrict__ idx) {
    const int b = blockIdx.x >> 2;
    const int s = ((blockIdx.x & 3) << 8) + threadIdx.x;   // 0..1023
    __shared__ float sx[NPTS * 3];
    const float* src = xyz + (size_t)b * NPTS * 3;
    for (int i = threadIdx.x; i < NPTS * 3; i += 256) sx[i] = src[i];
    __syncthreads();

    const float* q = new_xyz + ((size_t)b * NPOINT + s) * 3;
    const float qx = q[0], qy = q[1], qz = q[2];
    int* out = idx + ((size_t)b * NPOINT + s) * NSAMPLE;

    int cnt = 0, first = 0;
    for (int n = 0; n < NPTS && cnt < NSAMPLE; ++n) {
        const float dx = sx[n * 3 + 0] - qx;
        const float dy = sx[n * 3 + 1] - qy;
        const float dz = sx[n * 3 + 2] - qz;
        if (dx * dx + dy * dy + dz * dz <= RADIUS2) {
            if (cnt == 0) first = n;
            out[cnt++] = n;
        }
    }
    for (int j = cnt; j < NSAMPLE; ++j) out[j] = first;   // first==0 when cnt==0
}

// ---------------------------------------------------------------------------
// Gather + center + concat -> X rows (f16, stride KPAD, zero-padded 67..95)
__global__ void build_kernel(const float* __restrict__ xyz,
                             const float* __restrict__ pts,
                             const float* __restrict__ new_xyz,
                             const int* __restrict__ idx,
                             f16_t* __restrict__ X) {
    const int row = blockIdx.x * 256 + threadIdx.x;
    if (row >= ROWS) return;
    const int s = (row >> 5) & (NPOINT - 1);
    const int b = row >> 15;
    const int pid = idx[row];
    const float* p = xyz + ((size_t)b * NPTS + pid) * 3;
    const float* q = new_xyz + ((size_t)b * NPOINT + s) * 3;
    f16_t* xr = X + (size_t)row * KPAD;
    xr[0] = (f16_t)(p[0] - q[0]);
    xr[1] = (f16_t)(p[1] - q[1]);
    xr[2] = (f16_t)(p[2] - q[2]);
    const float* f = pts + ((size_t)b * NPTS + pid) * DFEAT;
    #pragma unroll 8
    for (int c = 0; c < DFEAT; ++c) xr[3 + c] = (f16_t)f[c];
    #pragma unroll
    for (int c = 3 + DFEAT; c < KPAD; ++c) xr[c] = (f16_t)0.0f;
}

// ---------------------------------------------------------------------------
// WMMA GEMM: Y[rows, NOUT] = X[rows, KPAD(f16)] * W[NOUT, Kin]^T + bias,
// plus per-channel sum / sum-of-squares accumulation for batchnorm.
// 256 threads = 8 waves; each wave owns one 16-row M tile.
// KT/NOUT are compile-time so every fragment lives in statically-named VGPRs
// (no v_movrels / v_nop padding) and all KT*NOUT/16 WMMAs unroll flat.
template <int KT, int NOUT>
__global__ void mlp_gemm_kernel(const f16_t* __restrict__ X,
                                const float* __restrict__ W,
                                const float* __restrict__ bias,
                                f16_t* __restrict__ Y,
                                float* __restrict__ stats,
                                int Kin) {
    __shared__ __align__(16) f16_t Wlds[128 * KPAD];   // 24 KB
    __shared__ float sStat[256];                       // sum[0..127], sq[128..255]

    const int tid = threadIdx.x;
    for (int i = tid; i < NOUT * KPAD; i += 256) {
        const int n = i / KPAD, c = i - n * KPAD;
        Wlds[i] = (c < Kin) ? (f16_t)W[n * Kin + c] : (f16_t)0.0f;
    }
    for (int i = tid; i < 256; i += 256) sStat[i] = 0.0f;
    __syncthreads();

    const int wave  = tid >> 5;
    const int lane  = tid & 31;
    const int ln15  = lane & 15;
    const int kb    = (lane >> 4) * 8;
    const int mtile = blockIdx.x * 8 + wave;
    const int rowBase = mtile * 16;

    // Prefetch the next M tile's A rows while we compute this one.
    __builtin_prefetch(X + ((size_t)(rowBase + 16 + ln15)) * KPAD + kb, 0, 1);

    // A fragments for all K tiles (wave32 16-bit A layout: two 16B runs per lane)
    v16h afr[KT];
    #pragma unroll
    for (int kt = 0; kt < KT; ++kt) {
        const f16_t* p = X + ((size_t)(rowBase + ln15)) * KPAD + kt * 32 + kb;
        const v8h lo = *(const v8h*)p;
        const v8h hi = *(const v8h*)(p + 16);
        v16h a;
        #pragma unroll
        for (int i = 0; i < 8; ++i) { a[i] = lo[i]; a[i + 8] = hi[i]; }
        afr[kt] = a;
    }

    const int mhalf = (lane < 16) ? 0 : 8;
    #pragma unroll
    for (int nt = 0; nt < (NOUT >> 4); ++nt) {
        v8f acc = {};
        #pragma unroll
        for (int kt = 0; kt < KT; ++kt) {
            const f16_t* q = &Wlds[(nt * 16 + ln15) * KPAD + kt * 32 + kb];
            const v8h lo = *(const v8h*)q;
            const v8h hi = *(const v8h*)(q + 16);
            v16h bf;
            #pragma unroll
            for (int i = 0; i < 8; ++i) { bf[i] = lo[i]; bf[i + 8] = hi[i]; }
            acc = __builtin_amdgcn_wmma_f32_16x16x32_f16(
                false, afr[kt], false, bf, (short)0, acc, false, false);
        }
        const int ch = nt * 16 + ln15;
        const float bv = bias[ch];
        float s = 0.0f, sq = 0.0f;
        #pragma unroll
        for (int r = 0; r < 8; ++r) {
            const float v = acc[r] + bv;
            s += v; sq += v * v;
            Y[(size_t)(rowBase + mhalf + r) * NOUT + ch] = (f16_t)v;
        }
        atomicAdd(&sStat[ch], s);
        atomicAdd(&sStat[128 + ch], sq);
    }
    __syncthreads();
    for (int i = tid; i < 2 * NOUT; i += 256) {
        const float v = (i < NOUT) ? sStat[i] : sStat[128 + (i - NOUT)];
        atomicAdd(&stats[i], v);
    }
}

// ---------------------------------------------------------------------------
// Fold BN stats + gamma/beta into y*sA + sB
__global__ void bnfin_kernel(const float* __restrict__ stats,
                             const float* __restrict__ gamma,
                             const float* __restrict__ beta,
                             float* __restrict__ par, int Nout) {
    const int c = threadIdx.x;
    if (c < Nout) {
        const float cnt  = (float)ROWS;
        const float mean = stats[c] / cnt;
        const float var  = stats[Nout + c] / cnt - mean * mean;
        const float sA   = gamma[c] * rsqrtf(var + BN_EPS);
        par[c]       = sA;
        par[128 + c] = beta[c] - mean * sA;
    }
}

// ---------------------------------------------------------------------------
// BN + ReLU, repack into next-layer X (stride KPAD, zero pad)
__global__ void bnrelu_kernel(const f16_t* __restrict__ Y,
                              const float* __restrict__ par,
                              f16_t* __restrict__ X) {
    const int row = blockIdx.x * 256 + threadIdx.x;
    if (row >= ROWS) return;
    const f16_t* y = Y + (size_t)row * 64;
    f16_t* xr = X + (size_t)row * KPAD;
    #pragma unroll 8
    for (int c = 0; c < 64; ++c) {
        const float v = (float)y[c] * par[c] + par[128 + c];
        xr[c] = (f16_t)fmaxf(v, 0.0f);
    }
    #pragma unroll
    for (int c = 64; c < KPAD; ++c) xr[c] = (f16_t)0.0f;
}

// ---------------------------------------------------------------------------
// Final: BN + ReLU on 128-ch layer, max over NSAMPLE -> d_out new_points
__global__ void maxpool_kernel(const f16_t* __restrict__ Y,
                               const float* __restrict__ par,
                               float* __restrict__ out) {
    const int g = blockIdx.x;        // b*NPOINT + s
    const int c = threadIdx.x;       // 0..127
    const f16_t* y = Y + (size_t)g * NSAMPLE * 128 + c;
    const float sA = par[c], sB = par[128 + c];
    float mx = -1e30f;
    #pragma unroll 8
    for (int k = 0; k < NSAMPLE; ++k) {
        const float v = fmaxf((float)y[k * 128] * sA + sB, 0.0f);
        mx = fmaxf(mx, v);
    }
    out[(size_t)g * 128 + c] = mx;
}

// ---------------------------------------------------------------------------
extern "C" void kernel_launch(void* const* d_in, const int* in_sizes, int n_in,
                              void* d_out, int out_size, void* d_ws, size_t ws_size,
                              hipStream_t stream) {
    const float* xyz = (const float*)d_in[0];
    const float* pts = (const float*)d_in[1];
    const float* Wl[3]  = { (const float*)d_in[2],  (const float*)d_in[6],  (const float*)d_in[10] };
    const float* Bl[3]  = { (const float*)d_in[3],  (const float*)d_in[7],  (const float*)d_in[11] };
    const float* Gl[3]  = { (const float*)d_in[4],  (const float*)d_in[8],  (const float*)d_in[12] };
    const float* Be[3]  = { (const float*)d_in[5],  (const float*)d_in[9],  (const float*)d_in[13] };

    float* out_xyz = (float*)d_out;                               // (16,1024,3)
    float* out_pts = (float*)d_out + (size_t)BATCH * NPOINT * 3;  // (16,1024,128)

    char*  ws      = (char*)d_ws;
    float* stats   = (float*)(ws + STATS_OFF);   // per layer: stats + l*256
    float* par     = (float*)(ws + PAR_OFF);     // per layer: par + l*256
    int*   fps_idx = (int*)  (ws + FPS_OFF);
    int*   idx     = (int*)  (ws + IDX_OFF);
    f16_t* X       = (f16_t*)(ws + X_OFF);
    f16_t* Y       = (f16_t*)(ws + Y_OFF);

    init_stats_kernel<<<1, 256, 0, stream>>>(stats);
    fps_kernel<<<BATCH, 256, 0, stream>>>(xyz, fps_idx, out_xyz);
    ball_kernel<<<BATCH * 4, 256, 0, stream>>>(xyz, out_xyz, idx);
    build_kernel<<<ROWS / 256, 256, 0, stream>>>(xyz, pts, out_xyz, idx, X);

    // layer 0: 67 -> 64   (3 k-tiles)
    mlp_gemm_kernel<3, 64><<<GEMM_BLOCKS, 256, 0, stream>>>(X, Wl[0], Bl[0], Y, stats + 0 * 256, 67);
    bnfin_kernel<<<1, 128, 0, stream>>>(stats + 0 * 256, Gl[0], Be[0], par + 0 * 256, 64);
    bnrelu_kernel<<<ROWS / 256, 256, 0, stream>>>(Y, par + 0 * 256, X);

    // layer 1: 64 -> 64   (2 k-tiles)
    mlp_gemm_kernel<2, 64><<<GEMM_BLOCKS, 256, 0, stream>>>(X, Wl[1], Bl[1], Y, stats + 1 * 256, 64);
    bnfin_kernel<<<1, 128, 0, stream>>>(stats + 1 * 256, Gl[1], Be[1], par + 1 * 256, 64);
    bnrelu_kernel<<<ROWS / 256, 256, 0, stream>>>(Y, par + 1 * 256, X);

    // layer 2: 64 -> 128  (2 k-tiles)
    mlp_gemm_kernel<2, 128><<<GEMM_BLOCKS, 256, 0, stream>>>(X, Wl[2], Bl[2], Y, stats + 2 * 256, 64);
    bnfin_kernel<<<1, 128, 0, stream>>>(stats + 2 * 256, Gl[2], Be[2], par + 2 * 256, 128);

    // max over samples -> new_points
    maxpool_kernel<<<BATCH * NPOINT, 128, 0, stream>>>(Y, par + 2 * 256, out_pts);
}
